// message_9285719294122
// MI455X (gfx1250) — matrix operand order repaired and checked by reading, not compiled, not run on hardware
//
#include <hip/hip_runtime.h>
#include <math.h>

// PaiNN-style fused edge message kernel for MI455X (gfx1250).
// E=262144 edges, F=128. GEMMs run as bf16 WMMA (f32 accumulate) to hit the
// HBM roofline (~540MB traffic dominates). Deterministic two-stage reduction
// through d_ws (needs 1MB).
//
// v3: GEMM1 computed transposed (D^T = W1^T x s^T) so its C-layout output is
// exactly the A-fragment layout GEMM2 needs -> phi1 never touches LDS (no
// transpose round-trip, -70KB LDS). RBF sines split across lane halves.

#define E_TOT   262144
#define F_DIM   128
#define NWG     512       // workgroups
#define WGS     512       // threads per workgroup (16 waves)
#define NWAVES  16
#define TPW     2         // edge tiles (of 256) per workgroup
#define TILE_E  256
#define SW      136       // LDS stride (halves) for K=128 matrices (+8 pad)
#define SR      40        // LDS stride (halves) for K=32 rbf matrices (+8 pad)
#define PI_F    3.14159265358979323846f

typedef __attribute__((ext_vector_type(16))) __bf16 v16bf;
typedef __attribute__((ext_vector_type(8)))  __bf16 v8bf;
typedef __attribute__((ext_vector_type(8)))  float  v8f;
typedef __attribute__((ext_vector_type(4)))  float  v4f;

__device__ __forceinline__ v8f wmma_bf16(v16bf a, v16bf b, v8f c) {
  // D = A(16x32 bf16) * B(32x16 bf16) + C(16x16 f32)
  return __builtin_amdgcn_wmma_f32_16x16x32_bf16(
      false, a, false, b, (short)0, c, false, false);
}

// B fragment: lane holds one column; two 8-half runs at K=kb+8h and
// K=kb+16+8h (p = kb+8h) from column-major LDS.
__device__ __forceinline__ v16bf lds_bfrag(const __bf16* p) {
  v8bf lo = *(const v8bf*)(p);
  v8bf hi = *(const v8bf*)(p + 16);
  return __builtin_shufflevector(lo, hi, 0, 1, 2, 3, 4, 5, 6, 7,
                                 8, 9, 10, 11, 12, 13, 14, 15);
}

// A fragment from bf16 LDS row: two 8-half runs at +8h and +16+8h.
__device__ __forceinline__ v16bf mkA_lds(const __bf16* p0, const __bf16* p1) {
  v8bf lo = *(const v8bf*)(p0);
  v8bf hi = *(const v8bf*)(p1);
  return __builtin_shufflevector(lo, hi, 0, 1, 2, 3, 4, 5, 6, 7,
                                 8, 9, 10, 11, 12, 13, 14, 15);
}

// B fragment from 16 contiguous global fp32 (s^T column = one s row slice).
__device__ __forceinline__ v16bf mkB_f32(const float* p) {
  v4f a = *(const v4f*)(p);
  v4f b = *(const v4f*)(p + 4);
  v4f c = *(const v4f*)(p + 8);
  v4f d = *(const v4f*)(p + 12);
  v16bf r;
  r[0]  = (__bf16)a[0]; r[1]  = (__bf16)a[1]; r[2]  = (__bf16)a[2]; r[3]  = (__bf16)a[3];
  r[4]  = (__bf16)b[0]; r[5]  = (__bf16)b[1]; r[6]  = (__bf16)b[2]; r[7]  = (__bf16)b[3];
  r[8]  = (__bf16)c[0]; r[9]  = (__bf16)c[1]; r[10] = (__bf16)c[2]; r[11] = (__bf16)c[3];
  r[12] = (__bf16)d[0]; r[13] = (__bf16)d[1]; r[14] = (__bf16)d[2]; r[15] = (__bf16)d[3];
  return r;
}

__global__ __launch_bounds__(WGS) void painn_fused_kernel(
    const float* __restrict__ s, const float* __restrict__ r,
    const float* __restrict__ v, const float* __restrict__ W1,
    const float* __restrict__ b1, const float* __restrict__ W2,
    const float* __restrict__ b2, const float* __restrict__ Wr,
    const float* __restrict__ br, float* __restrict__ ws_out) {
  // LDS: weights (bf16, column-major, padded) + per-wave rbf/edge scratch.
  __shared__ __align__(16) __bf16 W1s[128 * SW];            //  34,816 B
  __shared__ __align__(16) __bf16 W2s[384 * SW];            // 104,448 B
  __shared__ __align__(16) __bf16 Wrs[384 * SR];            //  30,720 B
  __shared__ __align__(16) __bf16 rbfS[NWAVES * 16 * SR];   //  20,480 B
  __shared__ __align__(16) float  edgeS[NWAVES * 16 * 4];   //   4,096 B
  __shared__ __align__(16) float  outLDS[512];              //   2,048 B
  // total ~197 KB of the WGP's 320 KB -> 1 WG (16 waves) resident per WGP.

  const int tid  = threadIdx.x;
  const int wave = tid >> 5;
  const int lane = tid & 31;
  const int lrow = lane & 15;  // WMMA row (A) / column (B,C) owned by lane
  const int h    = lane >> 4;  // lane half

  // ---- Stage weights to LDS as bf16, column-major [n][k] ----
  for (int idx = tid; idx < 128 * 128; idx += WGS) {
    int k = idx >> 7, n = idx & 127;              // W1[k][n]
    W1s[n * SW + k] = (__bf16)W1[idx];
  }
  for (int idx = tid; idx < 128 * 384; idx += WGS) {
    int k = idx / 384, n = idx % 384;             // W2[k][n]
    W2s[n * SW + k] = (__bf16)W2[idx];
  }
  for (int idx = tid; idx < 384 * 32; idx += WGS) {
    int n = idx >> 5, k = idx & 31;               // Wr[k][n], pad K 20->32
    Wrs[n * SR + k] = (k < 20) ? (__bf16)Wr[k * 384 + n] : (__bf16)0.0f;
  }
  for (int i = tid; i < 512; i += WGS) outLDS[i] = 0.0f;
  __syncthreads();

  float accv[3][8] = {};  // per-lane out_v partials: [dim][n-tile]
  float accs[8]    = {};  // per-lane out_s partials

  #pragma unroll 1
  for (int tt = 0; tt < TPW; ++tt) {
    const int tile  = blockIdx.x * TPW + tt;
    const int ebase = tile * TILE_E + wave * 16;   // this wave's 16 edges

    // ---- Per-edge scalars + RBF; both halves cooperate on one edge ----
    {
      const int e = ebase + lrow;
      float rx = r[e * 3 + 0], ry = r[e * 3 + 1], rz = r[e * 3 + 2];
      float dd  = sqrtf(rx * rx + ry * ry + rz * rz);
      float inv = 1.0f / dd;
      if (h == 0) {
        float fc = (dd < 5.0f) ? 0.5f * (__cosf(PI_F * dd * 0.2f) + 1.0f)
                               : 0.0f;
        float* es = &edgeS[(wave * 16 + lrow) * 4];
        es[0] = fc; es[1] = rx * inv; es[2] = ry * inv; es[3] = rz * inv;
      }
      __bf16* rb = &rbfS[(wave * 16 + lrow) * SR];
      const float w0 = PI_F * 0.2f * dd;
      #pragma unroll
      for (int n_ = 0; n_ < 10; ++n_) {  // h=0: n=1..10, h=1: n=11..20
        int k = 10 * h + n_;
        rb[k] = (__bf16)(__sinf(w0 * (float)(k + 1)) * inv);
      }
      #pragma unroll
      for (int n_ = 0; n_ < 6; ++n_)     // zero-pad K 20..31 split 6/6
        rb[20 + 6 * h + n_] = (__bf16)0.0f;
    }
    asm volatile("s_wait_dscnt 0" ::: "memory");  // same-wave LDS RAW

    // fcut for this lane's 8 C-rows (m = g + 8h); unit_r re-read from LDS
    // later only in the wv2 tiles (saves VGPRs).
    float fcut_g[8];
    #pragma unroll
    for (int g = 0; g < 8; ++g)
      fcut_g[g] = edgeS[(wave * 16 + g + 8 * h) * 4];

    // RBF A fragment (single K=32 chunk; A row = edge lrow).
    const __bf16* rp = &rbfS[(wave * 16 + lrow) * SR];
    v16bf rbfA = mkA_lds(rp + 8 * h, rp + 16 + 8 * h);

    // ---- GEMM1 (transposed): D^T = W1^T x s^T, eight 16(f)x16(edge) tiles.
    // A = W1^T tile (LDS, row f=16*t1+lrow), B = s^T (global, col=edge lrow,
    // 16 contiguous floats per chunk/half). C-layout of D^T: lane owns edge
    // column lrow and features f = 16*t1 + g + 8h -> exactly the GEMM2
    // A-fragment distribution, so phi never touches LDS. ----
    v8f acc1[8];
    #pragma unroll
    for (int t = 0; t < 8; ++t) acc1[t] = (v8f){};
    const float* srow = s + (size_t)(ebase + lrow) * 128;
    auto w1a = [&](int c, int t1) {  // A-frag of W1^T tile t1, chunk c
      const __bf16* base = &W1s[(t1 * 16 + lrow) * SW + 32 * c];
      return mkA_lds(base + 16 * h, base + 16 * h + 8);
    };
    v16bf bS = mkB_f32(srow + 16 * h);         // s^T chunk 0
    v16bf aW = w1a(0, 0);
    #pragma unroll
    for (int c = 0; c < 4; ++c) {
      v16bf bN = (c < 3) ? mkB_f32(srow + 32 * (c + 1) + 16 * h) : bS;
      #pragma unroll
      for (int t1 = 0; t1 < 8; ++t1) {
        const bool last = (c == 3) && (t1 == 7);
        v16bf aN = last ? aW : ((t1 < 7) ? w1a(c, t1 + 1) : w1a(c + 1, 0));
        acc1[t1] = wmma_bf16(aW, bS, acc1[t1]);
        aW = aN;
      }
      bS = bN;
    }

    // Bias + SiLU + pack GEMM2 A-fragments entirely in registers.
    v16bf phiA[4];
    #pragma unroll
    for (int c = 0; c < 4; ++c) {
      v16bf f;
      #pragma unroll
      for (int half = 0; half < 2; ++half) {
        const int t1 = 2 * c + half;
        const float* bp = b1 + 16 * t1 + 8 * h;  // 8 consecutive b1 values
        v4f bv0 = *(const v4f*)(bp);
        v4f bv1 = *(const v4f*)(bp + 4);
        #pragma unroll
        for (int j = 0; j < 4; ++j) {
          float x = acc1[t1][j] + bv0[j];
          f[8 * half + j] = (__bf16)(x / (1.0f + __expf(-x)));
        }
        #pragma unroll
        for (int j = 0; j < 4; ++j) {
          float x = acc1[t1][4 + j] + bv1[j];
          f[8 * half + 4 + j] = (__bf16)(x / (1.0f + __expf(-x)));
        }
      }
      phiA[c] = f;
    }

    // ---- GEMM2 + RBF GEMM over 24 output n-tiles, interleaved so that a
    // v-consuming tile (t<8) is followed by an s-tile and a w-tile, giving
    // the prefetched v loads ~10 WMMAs of latency cover. ----
    constexpr int seq[24] = {0, 8, 16, 1, 9, 17, 2,  10, 18, 3,  11, 19,
                             4, 12, 20, 5, 13, 21, 6, 14, 22, 7,  15, 23};
    auto w2p = [&](int t, int c) {
      return &W2s[(t * 16 + lrow) * SW + 32 * c + 8 * h];
    };
    auto wrp = [&](int t) { return &Wrs[(t * 16 + lrow) * SR + 8 * h]; };

    // Prefetch fragments + biases for seq[0] = 0.
    v16bf w2f[4], wrf;
    #pragma unroll
    for (int c = 0; c < 4; ++c) w2f[c] = lds_bfrag(w2p(0, c));
    wrf = lds_bfrag(wrp(0));
    float bb2 = b2[lrow], bbr = br[lrow];

    // Prefetch v values for v-tile 0.
    float vb[3][8];
    #pragma unroll
    for (int g = 0; g < 8; ++g) {
      const float* vp = v + (size_t)(ebase + g + 8 * h) * 384 + lrow;
      vb[0][g] = vp[0]; vb[1][g] = vp[128]; vb[2][g] = vp[256];
    }

    #pragma unroll
    for (int i = 0; i < 24; ++i) {
      const int t = seq[i];
      // Prefetch next tile's fragments + biases (LDS dscnt overlaps WMMA).
      v16bf nw2[4], nwr;
      float nb2, nbr;
      if (i < 23) {
        const int tn = seq[i + 1];
        #pragma unroll
        for (int c = 0; c < 4; ++c) nw2[c] = lds_bfrag(w2p(tn, c));
        nwr = lds_bfrag(wrp(tn));
        nb2 = b2[tn * 16 + lrow];
        nbr = br[tn * 16 + lrow];
      } else {
        #pragma unroll
        for (int c = 0; c < 4; ++c) nw2[c] = w2f[c];
        nwr = wrf; nb2 = bb2; nbr = bbr;
      }

      v8f acc2 = (v8f){};
      #pragma unroll
      for (int c = 0; c < 4; ++c) acc2 = wmma_bf16(phiA[c], w2f[c], acc2);
      v8f accW = (v8f){};
      accW = wmma_bf16(rbfA, wrf, accW);

      if (t < 8) {  // wv1 block: accumulate wv1[e,f] * v[e,d,f]
        #pragma unroll
        for (int g = 0; g < 8; ++g) {
          float split = (accW[g] + bbr) * fcut_g[g] * (acc2[g] + bb2);
          accv[0][t] = fmaf(split, vb[0][g], accv[0][t]);
          accv[1][t] = fmaf(split, vb[1][g], accv[1][t]);
          accv[2][t] = fmaf(split, vb[2][g], accv[2][t]);
        }
        if (t < 7) {  // prefetch v for the next v-tile (used 3 iters later)
          #pragma unroll
          for (int g = 0; g < 8; ++g) {
            const float* vp =
                v + (size_t)(ebase + g + 8 * h) * 384 + (t + 1) * 16 + lrow;
            vb[0][g] = vp[0]; vb[1][g] = vp[128]; vb[2][g] = vp[256];
          }
        }
      } else if (t < 16) {  // ws block: accumulate out_s
        #pragma unroll
        for (int g = 0; g < 8; ++g) {
          float split = (accW[g] + bbr) * fcut_g[g] * (acc2[g] + bb2);
          accs[t - 8] += split;
        }
      } else {  // wv2 block: accumulate unit_r[e,d] * wv2[e,f]
        #pragma unroll
        for (int g = 0; g < 8; ++g) {
          float split = (accW[g] + bbr) * fcut_g[g] * (acc2[g] + bb2);
          const float* es = &edgeS[(wave * 16 + g + 8 * h) * 4];
          accv[0][t - 16] = fmaf(es[1], split, accv[0][t - 16]);
          accv[1][t - 16] = fmaf(es[2], split, accv[1][t - 16]);
          accv[2][t - 16] = fmaf(es[3], split, accv[2][t - 16]);
        }
      }

      #pragma unroll
      for (int c = 0; c < 4; ++c) w2f[c] = nw2[c];
      wrf = nwr; bb2 = nb2; bbr = nbr;
    }
  }

  // ---- Workgroup reduction in LDS, then spill per-WG partials to d_ws ----
  #pragma unroll
  for (int t = 0; t < 8; ++t) {
    atomicAdd(&outLDS[0 * 128 + t * 16 + lrow], accv[0][t]);
    atomicAdd(&outLDS[1 * 128 + t * 16 + lrow], accv[1][t]);
    atomicAdd(&outLDS[2 * 128 + t * 16 + lrow], accv[2][t]);
    atomicAdd(&outLDS[384     + t * 16 + lrow], accs[t]);
  }
  __syncthreads();
  float* wsBase = ws_out + (size_t)blockIdx.x * 512;
  for (int i = tid; i < 512; i += WGS) wsBase[i] = outLDS[i];
}

// Deterministic final reduction over the 512 per-WG partials.
__global__ __launch_bounds__(512) void painn_reduce_kernel(
    const float* __restrict__ ws, float* __restrict__ out) {
  const int j = threadIdx.x;  // 512 output slots
  float a = 0.0f;
  for (int b = 0; b < NWG; ++b) a += ws[(size_t)b * 512 + j];
  out[j] = a;  // [0..383]=out_v (3,128), [384..511]=out_s
}

extern "C" void kernel_launch(void* const* d_in, const int* in_sizes, int n_in,
                              void* d_out, int out_size, void* d_ws,
                              size_t ws_size, hipStream_t stream) {
  (void)in_sizes; (void)n_in; (void)out_size; (void)ws_size;
  const float* s  = (const float*)d_in[0];
  const float* r  = (const float*)d_in[1];
  const float* v  = (const float*)d_in[2];
  const float* W1 = (const float*)d_in[3];
  const float* b1 = (const float*)d_in[4];
  const float* W2 = (const float*)d_in[5];
  const float* b2 = (const float*)d_in[6];
  const float* Wr = (const float*)d_in[7];
  const float* br = (const float*)d_in[8];
  float* out = (float*)d_out;
  float* ws  = (float*)d_ws;  // needs NWG*512*4 = 1 MB

  painn_fused_kernel<<<dim3(NWG), dim3(WGS), 0, stream>>>(
      s, r, v, W1, b1, W2, b2, Wr, br, ws);
  painn_reduce_kernel<<<dim3(1), dim3(512), 0, stream>>>(ws, out);
}
